// S4DConv_28449863368923
// MI455X (gfx1250) — matrix operand order repaired and checked by reading, not compiled, not run on hardware
//
#include <hip/hip_runtime.h>

// ---------------------------------------------------------------------------
// S4D chunked-scan on gfx1250 (wave32, WMMA bf16 16x16x32, f32 accumulate)
//
// B=8, L=2048, H=256, N=64, CH=1, T=256. Chunk Q=64, NC=32 chunks,
// BC = B*NC = 256 GEMM columns per h.
//
// Per h:  Y[64 x 256]  = T(Toeplitz k)      @ U          (intra-chunk)
//                      + P_re               @ S_re       (state read-out)
//                      + (-P_im)            @ S_im
//         X[64 x 256]  = Vin_{re,im}        @ U          (state injection)
//         S: 32-step scan  s <- A^Q s + X   (f32, per (h,n,b))
// All GEMM operands are pre-packed into the exact WMMA per-lane layout.
// ---------------------------------------------------------------------------

typedef __attribute__((ext_vector_type(16))) __bf16 v16bf;
typedef __attribute__((ext_vector_type(8)))  float  v8f;

#define S4D_B   8
#define S4D_L   2048
#define S4D_H   256
#define S4D_N   64
#define S4D_T   256
#define S4D_Q   64
#define S4D_NC  32
#define S4D_BC  256   // B * NC

__device__ __forceinline__ v8f s4d_wmma(v16bf a, v16bf b, v8f c) {
  // D = A(16x32 bf16) * B(32x16 bf16) + C(16x16 f32)
  return __builtin_amdgcn_wmma_f32_16x16x32_bf16(
      /*neg_a=*/false, a, /*neg_b=*/false, b,
      /*c_mod=*/(short)0, c, /*reuse_a=*/false, /*reuse_b=*/false);
}

// --- Kernel 1: per-(h,n) SSM parameters: dtA, A^Q, Cs2 = 2*C*(e^{dtA}-1)/w --
__global__ void s4d_params(const float* __restrict__ w_re, const float* __restrict__ w_im,
                           const float* __restrict__ C_re, const float* __restrict__ C_im,
                           const float* __restrict__ dt,
                           float2* __restrict__ dtA, float2* __restrict__ AQ,
                           float2* __restrict__ Cs2) {
  int tid = blockIdx.x * blockDim.x + threadIdx.x;   // 0 .. H*N-1
  int n = tid & (S4D_N - 1);
  int h = tid >> 6;
  int wrow = h % S4D_T;                              // tile 't n -> (v t) n'
  float wr = w_re[wrow * S4D_N + n];
  float wi = w_im[wrow * S4D_N + n];
  float d  = dt[h];
  float dar = wr * d, dai = wi * d;
  dtA[tid] = make_float2(dar, dai);
  float er = expf(dar), sn, cs;
  sincosf(dai, &sn, &cs);
  float A1r = er * cs, A1i = er * sn;                // A = exp(dtA)
  float erq = expf((float)S4D_Q * dar), snq, csq;
  sincosf((float)S4D_Q * dai, &snq, &csq);
  AQ[tid] = make_float2(erq * csq, erq * snq);       // A^Q
  float numr = A1r - 1.f, numi = A1i;                // (e^{dtA}-1)/w
  float inv = 1.f / (wr * wr + wi * wi);
  float qr = (numr * wr + numi * wi) * inv;
  float qi = (numi * wr - numr * wi) * inv;
  float cr = C_re[h * S4D_N + n], ci = C_im[h * S4D_N + n];  // CH==1
  Cs2[tid] = make_float2(2.f * (cr * qr - ci * qi),
                         2.f * (cr * qi + ci * qr));
}

// --- Kernel 2: short real kernel kq[h][j] = Re sum_n Cs2 * A^j, j<Q ---------
__global__ void s4d_kq(const float2* __restrict__ dtA, const float2* __restrict__ Cs2,
                       float* __restrict__ kq) {
  int h = blockIdx.x;
  int j = threadIdx.x;                               // 0..63
  float acc = 0.f;
  for (int n = 0; n < S4D_N; ++n) {
    float2 da = dtA[h * S4D_N + n];
    float2 c2 = Cs2[h * S4D_N + n];
    float er = expf((float)j * da.x), sn, cs;
    sincosf((float)j * da.y, &sn, &cs);
    acc += c2.x * (er * cs) - c2.y * (er * sn);
  }
  kq[h * S4D_Q + j] = acc;
}

// --- Kernel 3: pack the five 64x64 A-operands into WMMA A-layout (bf16) -----
// A-layout (16-bit A 16x32): lane: M = lane%16, half = lane/16;
// element e: K = e + 16*half (contiguous k, same order as the B packing).
__global__ void s4d_packA(const float* __restrict__ kq,
                          const float2* __restrict__ dtA, const float2* __restrict__ Cs2,
                          __bf16* __restrict__ packT,
                          __bf16* __restrict__ packPRe, __bf16* __restrict__ packPImN,
                          __bf16* __restrict__ packVinRe, __bf16* __restrict__ packVinIm) {
  int h  = blockIdx.x / 5;
  int m5 = blockIdx.x % 5;
  for (int it = 0; it < 16; ++it) {
    int idx  = it * 256 + threadIdx.x;               // 0..4095 within (h,matrix)
    int mt   = idx >> 10;
    int lane = (idx >> 4) & 31;
    int e    = idx & 15;
    int row  = mt * 16 + (lane & 15);                // M dim (t or n)
    int half = lane >> 4;
    int v = e >> 1, bb = e & 1;
    int kl = e + 16 * half;
    int k  = (((idx >> 9) & 1) << 5) + kl;           // K dim (j or n), 0..63
    float val;
    if (m5 == 0) {                                   // Toeplitz: T[t][j] = kq[t-j], j<=t
      val = (k <= row) ? kq[h * S4D_Q + (row - k)] : 0.f;
    } else if (m5 <= 2) {                            // P[t][n] = Cs2[n]*A[n]^{t+1}
      float2 da = dtA[h * S4D_N + k];
      float2 c2 = Cs2[h * S4D_N + k];
      float p = (float)(row + 1);
      float er = expf(p * da.x), sn, cs;
      sincosf(p * da.y, &sn, &cs);
      float pr = er * cs, pi = er * sn;
      val = (m5 == 1) ? (c2.x * pr - c2.y * pi)      // Re
                      : -(c2.x * pi + c2.y * pr);    // -Im (fold subtraction)
    } else {                                         // Vin[n][j] = A[n]^{Q-1-j}
      float2 da = dtA[h * S4D_N + row];
      float p = (float)(S4D_Q - 1 - k);
      float er = expf(p * da.x), sn, cs;
      sincosf(p * da.y, &sn, &cs);
      val = (m5 == 3) ? (er * cs) : (er * sn);
    }
    __bf16* dst = (m5 == 0) ? packT : (m5 == 1) ? packPRe : (m5 == 2) ? packPImN
                : (m5 == 3) ? packVinRe : packVinIm;
    dst[(size_t)h * 4096 + idx] = (__bf16)val;
  }
}

// --- Kernel 4: pack u into WMMA B-layout via LDS transpose ------------------
// B-layout (16-bit B 32x16): lane: N(col) = lane%16, half = lane/16;
// element e: K = e + 16*half.  Column index col = b*NC + c.
__global__ void s4d_packU(const float* __restrict__ u, __bf16* __restrict__ packU) {
  __shared__ float lds[S4D_Q][S4D_Q + 1];
  int bc = blockIdx.x;                               // column 0..255 -> (b,c)
  int b = bc >> 5, c = bc & 31;
  int h0 = blockIdx.y * 64;
  for (int it = 0; it < 16; ++it) {                  // coalesced along H
    int lin = it * 256 + threadIdx.x;
    int j = lin >> 6, hh = lin & 63;
    lds[j][hh] = u[((b * S4D_L) + (c * S4D_Q + j)) * S4D_H + h0 + hh];
  }
  __syncthreads();
  int hh  = threadIdx.x & 63;
  int grp = threadIdx.x >> 6;                        // 0..3 -> (kb,half)
  int h   = h0 + hh;
  int nt  = bc >> 4;
  int kb  = grp >> 1, half = grp & 1;
  int lane = (bc & 15) + 16 * half;
  size_t base = ((((size_t)h * 16 + nt) * 2 + kb) * 32 + lane) * 16;
  for (int e = 0; e < 16; ++e) {
    int j = grp * 16 + e;                            // K = kb*32 + e + 16*half
    packU[base + e] = (__bf16)lds[j][hh];
  }
}

// --- Kernel 5: X = Vin_{re,im} @ U   (WMMA, f32 out) ------------------------
__global__ void s4d_gemmX(const __bf16* __restrict__ packVinRe,
                          const __bf16* __restrict__ packVinIm,
                          const __bf16* __restrict__ packU,
                          float* __restrict__ XRe, float* __restrict__ XIm) {
  int wave = threadIdx.x >> 5, lane = threadIdx.x & 31;
  int gw = blockIdx.x * 8 + wave;
  int h = gw >> 4, nt = gw & 15;
  v8f accR[4] = {}, accI[4] = {};
#pragma unroll
  for (int kb = 0; kb < 2; ++kb) {
    v16bf bu = *(const v16bf*)(packU + ((((size_t)h * 16 + nt) * 2 + kb) * 32 + lane) * 16);
#pragma unroll
    for (int mt = 0; mt < 4; ++mt) {
      size_t aoff = ((((size_t)h * 4 + mt) * 2 + kb) * 32 + lane) * 16;
      v16bf ar = *(const v16bf*)(packVinRe + aoff);
      v16bf ai = *(const v16bf*)(packVinIm + aoff);
      accR[mt] = s4d_wmma(ar, bu, accR[mt]);
      accI[mt] = s4d_wmma(ai, bu, accI[mt]);
    }
  }
  int col = nt * 16 + (lane & 15), half = lane >> 4;
#pragma unroll
  for (int mt = 0; mt < 4; ++mt)
#pragma unroll
    for (int r = 0; r < 8; ++r) {
      int n = mt * 16 + r + 8 * half;
      XRe[((size_t)h * S4D_N + n) * S4D_BC + col] = accR[mt][r];
      XIm[((size_t)h * S4D_N + n) * S4D_BC + col] = accI[mt][r];
    }
}

// --- Kernel 6: 32-step scan s <- A^Q s + X; emit S in WMMA B-layout (bf16) --
__global__ void s4d_scan(const float2* __restrict__ AQ,
                         const float* __restrict__ XRe, const float* __restrict__ XIm,
                         __bf16* __restrict__ packSRe, __bf16* __restrict__ packSIm) {
  int tid = blockIdx.x * blockDim.x + threadIdx.x;   // H*N*B threads
  int b = tid & 7;
  int n = (tid >> 3) & 63;
  int h = tid >> 9;
  float2 aq = AQ[h * S4D_N + n];
  const float* xr = XRe + ((size_t)h * S4D_N + n) * S4D_BC + b * S4D_NC;
  const float* xi = XIm + ((size_t)h * S4D_N + n) * S4D_BC + b * S4D_NC;
  int kb = n >> 5, e = n & 15, lhi = n & 16;         // K = n in B-layout
  float sr = 0.f, si = 0.f;                          // state entering chunk 0 is 0
  for (int c = 0; c < S4D_NC; ++c) {
    int col = b * S4D_NC + c;
    int nt = col >> 4;
    int lane = (col & 15) + lhi;
    size_t off = ((((size_t)h * 16 + nt) * 2 + kb) * 32 + lane) * 16 + e;
    packSRe[off] = (__bf16)sr;                       // state *entering* chunk c
    packSIm[off] = (__bf16)si;
    float xrv = xr[c], xiv = xi[c];
    float nsr = aq.x * sr - aq.y * si + xrv;
    float nsi = aq.x * si + aq.y * sr + xiv;
    sr = nsr; si = nsi;
  }
}

// --- Kernel 7: Y = T@U + PRe@SRe + (-PIm)@SIm, + D*u skip, scatter to out ---
__global__ void s4d_gemmY(const __bf16* __restrict__ packT,
                          const __bf16* __restrict__ packPRe,
                          const __bf16* __restrict__ packPImN,
                          const __bf16* __restrict__ packU,
                          const __bf16* __restrict__ packSRe,
                          const __bf16* __restrict__ packSIm,
                          const float* __restrict__ u, const float* __restrict__ Dskip,
                          float* __restrict__ out) {
  int wave = threadIdx.x >> 5, lane = threadIdx.x & 31;
  int gw = blockIdx.x * 8 + wave;
  int h = gw >> 4, nt = gw & 15;
  size_t bbase = (((size_t)h * 16 + nt) * 2) * 32 * 16;
  __builtin_prefetch((const void*)(packSRe + bbase), 0, 1);  // global_prefetch_b8
  __builtin_prefetch((const void*)(packSIm + bbase), 0, 1);
  v8f acc[4] = {};
#pragma unroll
  for (int kb = 0; kb < 2; ++kb) {                   // intra-chunk Toeplitz
    v16bf bu = *(const v16bf*)(packU + bbase + (size_t)kb * 512 + (size_t)lane * 16);
#pragma unroll
    for (int mt = 0; mt < 4; ++mt) {
      v16bf a = *(const v16bf*)(packT + ((((size_t)h * 4 + mt) * 2 + kb) * 32 + lane) * 16);
      acc[mt] = s4d_wmma(a, bu, acc[mt]);
    }
  }
#pragma unroll
  for (int kb = 0; kb < 2; ++kb) {                   // + PRe @ SRe
    v16bf bs = *(const v16bf*)(packSRe + bbase + (size_t)kb * 512 + (size_t)lane * 16);
#pragma unroll
    for (int mt = 0; mt < 4; ++mt) {
      v16bf a = *(const v16bf*)(packPRe + ((((size_t)h * 4 + mt) * 2 + kb) * 32 + lane) * 16);
      acc[mt] = s4d_wmma(a, bs, acc[mt]);
    }
  }
#pragma unroll
  for (int kb = 0; kb < 2; ++kb) {                   // + (-PIm) @ SIm
    v16bf bs = *(const v16bf*)(packSIm + bbase + (size_t)kb * 512 + (size_t)lane * 16);
#pragma unroll
    for (int mt = 0; mt < 4; ++mt) {
      v16bf a = *(const v16bf*)(packPImN + ((((size_t)h * 4 + mt) * 2 + kb) * 32 + lane) * 16);
      acc[mt] = s4d_wmma(a, bs, acc[mt]);
    }
  }
  int col = nt * 16 + (lane & 15), half = lane >> 4;
  int b = col >> 5, c = col & 31;
  float dh = Dskip[h];                               // D[0][h], CH==1
#pragma unroll
  for (int mt = 0; mt < 4; ++mt)
#pragma unroll
    for (int r = 0; r < 8; ++r) {
      int t = mt * 16 + r + 8 * half;                // in-chunk time
      int l = c * S4D_Q + t;
      size_t gidx = ((size_t)b * S4D_L + l) * S4D_H + h;  // out[B,L,1,H]
      out[gidx] = acc[mt][r] + dh * u[gidx];
    }
}

// ---------------------------------------------------------------------------
extern "C" void kernel_launch(void* const* d_in, const int* in_sizes, int n_in,
                              void* d_out, int out_size, void* d_ws, size_t ws_size,
                              hipStream_t stream) {
  (void)in_sizes; (void)n_in; (void)out_size; (void)ws_size;
  const float* u    = (const float*)d_in[0];
  const float* w_re = (const float*)d_in[1];
  const float* w_im = (const float*)d_in[2];
  const float* C_re = (const float*)d_in[3];
  const float* C_im = (const float*)d_in[4];
  const float* Dsk  = (const float*)d_in[5];
  const float* dt   = (const float*)d_in[6];
  float* out = (float*)d_out;

  char* ws = (char*)d_ws;
  size_t off = 0;
  auto alloc = [&](size_t bytes) { char* p = ws + off; off = (off + bytes + 255) & ~(size_t)255; return p; };
  float2* dtA   = (float2*)alloc((size_t)S4D_H * S4D_N * sizeof(float2));
  float2* AQ    = (float2*)alloc((size_t)S4D_H * S4D_N * sizeof(float2));
  float2* Cs2   = (float2*)alloc((size_t)S4D_H * S4D_N * sizeof(float2));
  float*  kq    = (float*) alloc((size_t)S4D_H * S4D_Q * sizeof(float));
  __bf16* packT    = (__bf16*)alloc((size_t)S4D_H * 4096 * sizeof(__bf16));
  __bf16* packPRe  = (__bf16*)alloc((size_t)S4D_H * 4096 * sizeof(__bf16));
  __bf16* packPImN = (__bf16*)alloc((size_t)S4D_H * 4096 * sizeof(__bf16));
  __bf16* packVinR = (__bf16*)alloc((size_t)S4D_H * 4096 * sizeof(__bf16));
  __bf16* packVinI = (__bf16*)alloc((size_t)S4D_H * 4096 * sizeof(__bf16));
  __bf16* packU    = (__bf16*)alloc((size_t)S4D_H * 16384 * sizeof(__bf16));
  float*  XRe      = (float*) alloc((size_t)S4D_H * S4D_N * S4D_BC * sizeof(float));
  float*  XIm      = (float*) alloc((size_t)S4D_H * S4D_N * S4D_BC * sizeof(float));
  __bf16* packSRe  = (__bf16*)alloc((size_t)S4D_H * 16384 * sizeof(__bf16));
  __bf16* packSIm  = (__bf16*)alloc((size_t)S4D_H * 16384 * sizeof(__bf16));

  s4d_params<<<(S4D_H * S4D_N) / 256, 256, 0, stream>>>(w_re, w_im, C_re, C_im, dt,
                                                        dtA, AQ, Cs2);
  s4d_kq<<<S4D_H, S4D_Q, 0, stream>>>(dtA, Cs2, kq);
  s4d_packA<<<S4D_H * 5, 256, 0, stream>>>(kq, dtA, Cs2,
                                           packT, packPRe, packPImN, packVinR, packVinI);
  s4d_packU<<<dim3(S4D_BC, S4D_H / 64), 256, 0, stream>>>(u, packU);
  s4d_gemmX<<<(S4D_H * 16) / 8, 256, 0, stream>>>(packVinR, packVinI, packU, XRe, XIm);
  s4d_scan<<<(S4D_H * S4D_N * S4D_B) / 256, 256, 0, stream>>>(AQ, XRe, XIm, packSRe, packSIm);
  s4d_gemmY<<<(S4D_H * 16) / 8, 256, 0, stream>>>(packT, packPRe, packPImN,
                                                  packU, packSRe, packSIm, u, Dsk, out);
}
